// RGCN_21105469293025
// MI455X (gfx1250) — matrix operand order repaired and checked by reading, not compiled, not run on hardware
//
#include <hip/hip_runtime.h>
#include <hip/hip_bf16.h>

#define N_NODES 50000
#define N_EDGES 800000
#define DIMF    128
#define NREL    8
#define KAGG    (NREL * DIMF)      // 1024
#define KTOT    (KAGG + DIMF)      // 1152 (8 relation blocks + root block)

static_assert(N_NODES % 16 == 0, "M-tiling assumes N divisible by 16");

typedef __attribute__((ext_vector_type(16))) __bf16 v16bf;
typedef __attribute__((ext_vector_type(8)))  float  v8f;
typedef __attribute__((ext_vector_type(4)))  unsigned int v4u;
typedef __attribute__((ext_vector_type(8)))  int    v8i;
typedef __attribute__((ext_vector_type(4)))  int    v4i;

union BFrag { v16bf v; uint4 q[2]; };

#if defined(__HIP_DEVICE_COMPILE__) && __has_builtin(__builtin_amdgcn_tensor_load_to_lds)
#define USE_TDM 1
#else
#define USE_TDM 0
#endif

// ---------------------------------------------------------------------------
// Pack per-layer weights into Wt[l][h][k] (bf16, K-major rows => contiguous
// b128 loads for the WMMA B fragment, which wants column h, ascending k).
// k < 1024: W[r=k>>7][d=k&127][h]; k >= 1024: root[k-1024][h].
// ---------------------------------------------------------------------------
__global__ void prep_weights_kernel(const float* __restrict__ W0, const float* __restrict__ r0,
                                    const float* __restrict__ W1, const float* __restrict__ r1,
                                    const float* __restrict__ W2, const float* __restrict__ r2,
                                    __bf16* __restrict__ Wt) {
    int idx = blockIdx.x * blockDim.x + threadIdx.x;
    if (idx >= 3 * DIMF * KTOT) return;
    int l = idx / (DIMF * KTOT);
    int rem = idx - l * (DIMF * KTOT);
    int h = rem / KTOT;
    int k = rem - h * KTOT;
    const float* W  = (l == 0) ? W0 : (l == 1) ? W1 : W2;
    const float* rt = (l == 0) ? r0 : (l == 1) ? r1 : r2;
    float v = (k < KAGG) ? W[(size_t)k * DIMF + h] : rt[(size_t)(k - KAGG) * DIMF + h];
    Wt[idx] = (__bf16)v;
}

// ---------------------------------------------------------------------------
// Per-(dst, relation) in-degree counts, then reciprocal for mean-normalization
// ---------------------------------------------------------------------------
__global__ void count_edges_kernel(const int* __restrict__ dst, const int* __restrict__ et,
                                   int* __restrict__ cnt) {
    int e = blockIdx.x * blockDim.x + threadIdx.x;
    if (e < N_EDGES) atomicAdd(&cnt[dst[e] * NREL + et[e]], 1);
}

__global__ void inv_count_kernel(const int* __restrict__ cnt, float* __restrict__ invc) {
    int i = blockIdx.x * blockDim.x + threadIdx.x;
    if (i < N_NODES * NREL) {
        int c = cnt[i];
        invc[i] = 1.0f / (float)((c > 0) ? c : 1);
    }
}

// ---------------------------------------------------------------------------
// Aggregate-first scatter: one wave32 per edge, lane covers 4 features.
// x rows (25.6MB total) are L2-resident -> gathers hit L2; atomics hit agg.
// ---------------------------------------------------------------------------
__global__ __launch_bounds__(256)
void scatter_mean_kernel(const float* __restrict__ x,
                         const int* __restrict__ src, const int* __restrict__ dst,
                         const int* __restrict__ et, const float* __restrict__ invc,
                         float* __restrict__ agg) {
    int e = blockIdx.x * (blockDim.x >> 5) + (threadIdx.x >> 5);
    int lane = threadIdx.x & 31;
    if (e >= N_EDGES) return;
    int s = src[e], d = dst[e], r = et[e];
    float w = invc[d * NREL + r];
    const float4* xr = (const float4*)(x + (size_t)s * DIMF);
    float4 v = xr[lane];
    float* ap = agg + (size_t)d * KAGG + (size_t)r * DIMF + lane * 4;
    unsafeAtomicAdd(ap + 0, v.x * w);
    unsafeAtomicAdd(ap + 1, v.y * w);
    unsafeAtomicAdd(ap + 2, v.z * w);
    unsafeAtomicAdd(ap + 3, v.w * w);
}

#if USE_TDM
// ---------------------------------------------------------------------------
// TDM flat copy: nElems fp32 from global -> LDS, one 1-row tile
// (D# per CDNA5 ISA: group0 {count=1, lds_addr,
//  global_addr, type=2}; group1 {data_size=2 (4B), tensor_dim0 = tile_dim0 =
//  tensor_dim0_stride = nElems, tensor_dim1 = tile_dim1 = 1}).
// 6-arg builtin form (clang-23 / therock-10.0): groups 2/3 + trailing v8i
// are unused for a <=2D tile -> zero-filled; cpol=0.
// ---------------------------------------------------------------------------
__device__ __forceinline__ void tdm_load_flat_f32(const float* gsrc, unsigned int lds_byte_addr,
                                                  unsigned int nElems) {
    unsigned long long ga = (unsigned long long)(uintptr_t)gsrc;
    v4u g0;
    g0[0] = 1u;                                             // count=1 (valid), user D#
    g0[1] = lds_byte_addr;                                  // lds_addr
    g0[2] = (unsigned int)(ga & 0xFFFFFFFFu);               // global_addr[31:0]
    g0[3] = (unsigned int)((ga >> 32) & 0x1FFFFFFu)         // global_addr[56:32]
          | (2u << 30);                                     // type=2 ("image")
    v8i g1;
    g1[0] = (int)(2u << 16);                                // wg_mask=0, data_size=4B
    g1[1] = (int)((nElems & 0xFFFFu) << 16);                // tensor_dim0[15:0]
    g1[2] = (int)(((nElems >> 16) & 0xFFFFu) | (1u << 16)); // tensor_dim0[31:16] | tensor_dim1=1
    g1[3] = (int)((nElems & 0xFFFFu) << 16);                // tile_dim0
    g1[4] = 1;                                              // tile_dim1=1, tile_dim2=0
    g1[5] = (int)nElems;                                    // tensor_dim0_stride[31:0]
    g1[6] = 0;                                              // stride hi / dim1_stride lo
    g1[7] = 0;
    v4i z4 = {0, 0, 0, 0};                                  // groups 2/3 unused (<=2D)
    v8i z8 = {0, 0, 0, 0, 0, 0, 0, 0};
    __builtin_amdgcn_tensor_load_to_lds(g0, g1, z4, z4, z8, 0);
}
#endif

// ---------------------------------------------------------------------------
// Fused RGCN layer GEMM:  y[n, 0:128] = (agg[n, 0:1024] || x[n, 0:128]) * WtL^T
//                                       + bias (+ ReLU)
// Workgroup = 256 thr = 8 waves; block owns a 16-row strip. The fp32 A strip
// (16x1024 agg + 16x128 x = 72KB) is DMA'd into dynamic LDS by the Tensor
// Data Mover (wave 0 issues 2 TENSOR_LOAD_TO_LDS, s_wait_tensorcnt, barrier).
// Each wave owns one 16-column tile: 36 V_WMMA_F32_16X16X32_BF16 K-steps,
// converting fp32->bf16 at fragment build (co-executes with XDL WMMA).
// Fragment layouts follow CDNA5 ISA §7.12.2.
// ---------------------------------------------------------------------------
__device__ __forceinline__ v16bf make_afrag(const float* __restrict__ ar) {
    // ar points at A[m, k0 + kh*8]; frag = {k0+kh*8+0..7, k0+16+kh*8+0..7}
    float4 f0 = *(const float4*)(ar + 0);
    float4 f1 = *(const float4*)(ar + 4);
    float4 f2 = *(const float4*)(ar + 16);
    float4 f3 = *(const float4*)(ar + 20);
    v16bf a;
    a[0]  = (__bf16)f0.x; a[1]  = (__bf16)f0.y; a[2]  = (__bf16)f0.z; a[3]  = (__bf16)f0.w;
    a[4]  = (__bf16)f1.x; a[5]  = (__bf16)f1.y; a[6]  = (__bf16)f1.z; a[7]  = (__bf16)f1.w;
    a[8]  = (__bf16)f2.x; a[9]  = (__bf16)f2.y; a[10] = (__bf16)f2.z; a[11] = (__bf16)f2.w;
    a[12] = (__bf16)f3.x; a[13] = (__bf16)f3.y; a[14] = (__bf16)f3.z; a[15] = (__bf16)f3.w;
    return a;
}

__global__ __launch_bounds__(256)
void rgcn_gemm_kernel(const float* __restrict__ agg, const float* __restrict__ xin,
                      const __bf16* __restrict__ WtL, const float* __restrict__ bias,
                      float* __restrict__ y, int do_relu) {
    extern __shared__ float lds[];
    float* As1 = lds;                      // [16][1024] fp32 (relations part)
    float* As2 = lds + 16 * KAGG;          // [16][128]  fp32 (root/x part)

    const int tid = threadIdx.x;
    const int wv  = tid >> 5;
    const int rowBase = blockIdx.x * 16;

#if USE_TDM
    if (wv == 0) {
        tdm_load_flat_f32(agg + (size_t)rowBase * KAGG,
                          (unsigned int)(uintptr_t)As1, 16u * KAGG);
        tdm_load_flat_f32(xin + (size_t)rowBase * DIMF,
                          (unsigned int)(uintptr_t)As2, 16u * DIMF);
        __builtin_amdgcn_s_wait_tensorcnt(0);
    }
    __syncthreads();
#else
    // Fallback: cooperative fp32 copy (16*1152/4 = 4608 float4s / 256 thr)
    for (int it = 0; it < (16 * KTOT / 4) / 256; ++it) {
        int g  = it * 256 + tid;
        int r  = g / (KTOT / 4);
        int kg = (g - r * (KTOT / 4)) * 4;
        float4 v;
        float* dst;
        if (kg < KAGG) { v = *(const float4*)(agg + (size_t)(rowBase + r) * KAGG + kg);
                         dst = As1 + r * KAGG + kg; }
        else           { v = *(const float4*)(xin + (size_t)(rowBase + r) * DIMF + (kg - KAGG));
                         dst = As2 + r * DIMF + (kg - KAGG); }
        *(float4*)dst = v;
    }
    __syncthreads();
#endif

    const int lane = tid & 31;
    const int colBase = wv * 16;          // 8 waves cover all 128 output columns
    const int kh = lane >> 4;             // K-half select (ISA A/B lane split)
    const int ml = lane & 15;             // A: row M; B: column N

    const __bf16* Brow = WtL + (size_t)(colBase + ml) * KTOT;
    const float*  Arow1 = As1 + ml * KAGG + kh * 8;
    const float*  Arow2 = As2 + ml * DIMF + kh * 8;

    v8f c = {};
    // Relation blocks: K = 0 .. 1023
    for (int kb = 0; kb < KAGG / 32; ++kb) {
        const int k0 = kb * 32;
        v16bf a = make_afrag(Arow1 + k0);
        // B 32x16 bf16 frag: lane holds column ml, K = {k0+kh*8+0..7, k0+16+kh*8+0..7}
        BFrag b;
        const uint4* bp = (const uint4*)(Brow + k0 + kh * 8);
        b.q[0] = bp[0];
        b.q[1] = bp[2];
        __builtin_prefetch(Brow + k0 + 64, 0, 0);   // global_prefetch_b8
        c = __builtin_amdgcn_wmma_f32_16x16x32_bf16(
                false, a, false, b.v, (short)0, c, false, false);
    }
    // Root block: K = 1024 .. 1151
    for (int kb = 0; kb < DIMF / 32; ++kb) {
        const int k0 = kb * 32;
        v16bf a = make_afrag(Arow2 + k0);
        BFrag b;
        const uint4* bp = (const uint4*)(Brow + KAGG + k0 + kh * 8);
        b.q[0] = bp[0];
        b.q[1] = bp[2];
        c = __builtin_amdgcn_wmma_f32_16x16x32_bf16(
                false, a, false, b.v, (short)0, c, false, false);
    }

    // C/D layout: lane n = lane&15 => column; VGPR v => row v + 8*(lane>=16)
    const int col = colBase + ml;
    const float bv = bias[col];
    const int mhi = kh * 8;
    #pragma unroll
    for (int vI = 0; vI < 8; ++vI) {
        float o = c[vI] + bv;
        if (do_relu) o = fmaxf(o, 0.f);
        y[(size_t)(rowBase + mhi + vI) * DIMF + col] = o;
    }
}

// ---------------------------------------------------------------------------
// Column sums for the node-mean readout (divide by N in the head kernel)
// ---------------------------------------------------------------------------
#define ROWS_PER_BLK 512
__global__ void col_sum_kernel(const float* __restrict__ y, float* __restrict__ u) {
    int col = threadIdx.x;                 // 128 threads
    int r0 = blockIdx.x * ROWS_PER_BLK;
    int r1 = r0 + ROWS_PER_BLK; if (r1 > N_NODES) r1 = N_NODES;
    float s = 0.f;
    for (int r = r0; r < r1; ++r) s += y[(size_t)r * DIMF + col];
    unsafeAtomicAdd(&u[col], s);
}

// ---------------------------------------------------------------------------
// MLP head: relu(concat(u1,u2)/N @ fc1 + b1) @ fc2 + b2  ->  scalar
// ---------------------------------------------------------------------------
__global__ void head_kernel(const float* __restrict__ u,
                            const float* __restrict__ fc1w, const float* __restrict__ fc1b,
                            const float* __restrict__ fc2w, const float* __restrict__ fc2b,
                            float* __restrict__ out) {
    __shared__ float sh[128];
    int h = threadIdx.x;
    const float invN = 1.0f / (float)N_NODES;
    float acc = 0.f;
    for (int k = 0; k < 2 * DIMF; ++k)
        acc += (u[k] * invN) * fc1w[(size_t)k * DIMF + h];
    float t = fmaxf(acc + fc1b[h], 0.f);
    sh[h] = t * fc2w[h];
    __syncthreads();
    for (int s = 64; s > 0; s >>= 1) {
        if (h < s) sh[h] += sh[h + s];
        __syncthreads();
    }
    if (h == 0) out[0] = sh[0] + fc2b[0];
}

// ---------------------------------------------------------------------------
extern "C" void kernel_launch(void* const* d_in, const int* in_sizes, int n_in,
                              void* d_out, int out_size, void* d_ws, size_t ws_size,
                              hipStream_t stream) {
    (void)in_sizes; (void)n_in; (void)out_size; (void)ws_size;

    const float* x[2]   = { (const float*)d_in[0], (const float*)d_in[3] };
    const int*   ei[2]  = { (const int*)d_in[1],   (const int*)d_in[4] };
    const int*   et[2]  = { (const int*)d_in[2],   (const int*)d_in[5] };
    const float* W0 = (const float*)d_in[6],  *r0 = (const float*)d_in[7],  *b0 = (const float*)d_in[8];
    const float* W1 = (const float*)d_in[9],  *r1 = (const float*)d_in[10], *b1 = (const float*)d_in[11];
    const float* W2 = (const float*)d_in[12], *r2 = (const float*)d_in[13], *b2 = (const float*)d_in[14];
    const float* fc1w = (const float*)d_in[15];
    const float* fc1b = (const float*)d_in[16];
    const float* fc2w = (const float*)d_in[17];
    const float* fc2b = (const float*)d_in[18];
    const float* biasL[3] = { b0, b1, b2 };

    char* ws = (char*)d_ws;
    size_t off = 0;
    float*  agg  = (float*)(ws + off);  off += (size_t)N_NODES * KAGG * sizeof(float);   // 204.8 MB
    float*  yA   = (float*)(ws + off);  off += (size_t)N_NODES * DIMF * sizeof(float);   // 25.6 MB
    float*  yB   = (float*)(ws + off);  off += (size_t)N_NODES * DIMF * sizeof(float);   // 25.6 MB
    int*    cnt  = (int*)  (ws + off);  off += (size_t)N_NODES * NREL * sizeof(int);     // 1.6 MB
    float*  invc = (float*)(ws + off);  off += (size_t)N_NODES * NREL * sizeof(float);   // 1.6 MB
    __bf16* Wt   = (__bf16*)(ws + off); off += (size_t)3 * DIMF * KTOT * sizeof(__bf16); // 0.88 MB
    float*  u    = (float*)(ws + off);  off += 2 * DIMF * sizeof(float);

    // Pack weights (3 layers) into bf16 transposed form
    prep_weights_kernel<<<(3 * DIMF * KTOT + 255) / 256, 256, 0, stream>>>(W0, r0, W1, r1, W2, r2, Wt);
    (void)hipMemsetAsync(u, 0, 2 * DIMF * sizeof(float), stream);

    const size_t ldsBytes = (size_t)16 * KTOT * sizeof(float);   // 72 KB dynamic LDS

    for (int g = 0; g < 2; ++g) {
        const int* src = ei[g];
        const int* dst = ei[g] + N_EDGES;
        const int* ety = et[g];

        (void)hipMemsetAsync(cnt, 0, (size_t)N_NODES * NREL * sizeof(int), stream);
        count_edges_kernel<<<(N_EDGES + 255) / 256, 256, 0, stream>>>(dst, ety, cnt);
        inv_count_kernel<<<(N_NODES * NREL + 255) / 256, 256, 0, stream>>>(cnt, invc);

        const float* cur = x[g];
        for (int l = 0; l < 3; ++l) {
            (void)hipMemsetAsync(agg, 0, (size_t)N_NODES * KAGG * sizeof(float), stream);
            scatter_mean_kernel<<<(N_EDGES + 7) / 8, 256, 0, stream>>>(cur, src, dst, ety, invc, agg);
            float* out_l = (l & 1) ? yB : yA;
            rgcn_gemm_kernel<<<N_NODES / 16, 256, ldsBytes, stream>>>(
                agg, cur, Wt + (size_t)l * DIMF * KTOT, biasL[l], out_l, (l < 2) ? 1 : 0);
            cur = out_l;
        }
        col_sum_kernel<<<(N_NODES + ROWS_PER_BLK - 1) / ROWS_PER_BLK, 128, 0, stream>>>(cur, u + g * DIMF);
    }

    head_kernel<<<1, 128, 0, stream>>>(u, fc1w, fc1b, fc2w, fc2b, (float*)d_out);
}
